// PointGroup_91061896610561
// MI455X (gfx1250) — matrix operand order repaired and checked by reading, not compiled, not run on hardware
//
#include <hip/hip_runtime.h>
#include <math.h>

// ---------------------------------------------------------------------------
// PointGroup heads for MI455X (gfx1250, wave32, WMMA).
// Outputs concatenated in d_out (fp32):
//   [0, 20N)        semantic_scores [N,20]
//   [20N, 21N)      semantic_preds  [N]      (argmax index stored as float)
//   [21N, 24N)      pt_offsets      [N,3]
//   [24N, 24N+1024) proposal scores [1024]
// ---------------------------------------------------------------------------

typedef __attribute__((ext_vector_type(16))) _Float16 v16h;
typedef __attribute__((ext_vector_type(8)))  float    v8f;

#define BN_EPS   1e-4f
#define NPROP    1024
#define MFEAT    32

#define WMMA_F16(A, B, C) \
  __builtin_amdgcn_wmma_f32_16x16x32_f16(false, (A), false, (B), (short)0, (C), false, false)

// Build a 32x16 (KxN) f16 B-fragment from a row-major [32][ncols] fp32 weight
// matrix held in LDS. ISA layout: lanes 0-15 hold K=0..7,16..23, lanes 16-31
// hold K=8..15,24..31; element e -> K = kb + e (+8 if e >= 8); N = (lane&15)+noff.
__device__ __forceinline__ v16h make_bfrag(const float* W, int ncols, int noff, int lane) {
  int n  = (lane & 15) + noff;
  int kb = (lane >= 16) ? 8 : 0;
  v16h b;
#pragma unroll
  for (int e = 0; e < 16; ++e)
    b[e] = (n < ncols) ? (_Float16)W[(kb + e + ((e >= 8) ? 8 : 0)) * ncols + n] : (_Float16)0.0f;
  return b;
}

// Pack 4 float4 (K blocks base..base+7 and base+16..base+23) into an f16
// A-fragment per the 16-bit 16x32 A layout.
__device__ __forceinline__ v16h pack_afrag(float4 q0, float4 q1, float4 q2, float4 q3) {
  v16h a;
  a[0]=(_Float16)q0.x;  a[1]=(_Float16)q0.y;  a[2]=(_Float16)q0.z;  a[3]=(_Float16)q0.w;
  a[4]=(_Float16)q1.x;  a[5]=(_Float16)q1.y;  a[6]=(_Float16)q1.z;  a[7]=(_Float16)q1.w;
  a[8]=(_Float16)q2.x;  a[9]=(_Float16)q2.y;  a[10]=(_Float16)q2.z; a[11]=(_Float16)q2.w;
  a[12]=(_Float16)q3.x; a[13]=(_Float16)q3.y; a[14]=(_Float16)q3.z; a[15]=(_Float16)q3.w;
  return a;
}

// ---------------------------------------------------------------------------
// Kernel A: fused semantic + offset heads. One 16-point tile per wave per
// loop iteration; 5 WMMAs per tile. All global writes fully coalesced.
// Assumes N % 16 == 0 (true for N = 1e6).
// ---------------------------------------------------------------------------
__global__ void __launch_bounds__(256) pg_heads_kernel(
    const float* __restrict__ feats,
    const float* __restrict__ sem_w,  const float* __restrict__ sem_b,
    const float* __restrict__ off_w1, const float* __restrict__ off_b1,
    const float* __restrict__ bn_g,   const float* __restrict__ bn_b,
    const float* __restrict__ off_w2, const float* __restrict__ off_b2,
    float* __restrict__ out_scores, float* __restrict__ out_preds,
    float* __restrict__ out_offsets, int n_tiles)
{
  __shared__ float ws[32*20 + 32*32 + 32*3 + 20 + 32 + 32 + 32 + 4];
  __shared__ float slab_all[8][16 * 32];   // per-wave 2KB scratch tile

  float* Wsem = ws;
  float* W1   = Wsem + 32*20;
  float* W2   = W1   + 32*32;
  float* Bs   = W2   + 32*3;
  float* B1   = Bs   + 20;
  float* Gg   = B1   + 32;
  float* Bt   = Gg   + 32;
  float* B2   = Bt   + 32;

  for (int i = threadIdx.x; i < 32*20; i += 256) Wsem[i] = sem_w[i];
  for (int i = threadIdx.x; i < 32*32; i += 256) W1[i]   = off_w1[i];
  for (int i = threadIdx.x; i < 32*3;  i += 256) W2[i]   = off_w2[i];
  if (threadIdx.x < 20) Bs[threadIdx.x] = sem_b[threadIdx.x];
  if (threadIdx.x < 32) {
    B1[threadIdx.x] = off_b1[threadIdx.x];
    Gg[threadIdx.x] = bn_g[threadIdx.x];
    Bt[threadIdx.x] = bn_b[threadIdx.x];
  }
  if (threadIdx.x < 3) B2[threadIdx.x] = off_b2[threadIdx.x];
  __syncthreads();

  const int lane = threadIdx.x & 31;
  const int wid  = threadIdx.x >> 5;
  float* slab = slab_all[wid];
  const int n  = lane & 15;
  const int hi = (lane >= 16) ? 1 : 0;
  const int m0 = hi ? 8 : 0;

  // Weight fragments: built once, reused across all tiles of this wave.
  v16h bs0 = make_bfrag(Wsem, 20, 0,  lane);
  v16h bs1 = make_bfrag(Wsem, 20, 16, lane);
  v16h b10 = make_bfrag(W1,   32, 0,  lane);
  v16h b11 = make_bfrag(W1,   32, 16, lane);
  v16h b2  = make_bfrag(W2,    3, 0,  lane);

  const float rs  = rsqrtf(1.0f + BN_EPS);
  const float sb0 = Bs[n];
  const float sb1 = (n + 16 < 20) ? Bs[n + 16] : 0.0f;
  const float ob0 = B1[n],      gg0 = Gg[n]      * rs, bb0 = Bt[n];
  const float ob1 = B1[n + 16], gg1 = Gg[n + 16] * rs, bb1 = Bt[n + 16];
  const float ob2 = (n < 3) ? B2[n] : 0.0f;

  const int total_waves = (int)gridDim.x * 8;

  for (int tile = (int)blockIdx.x * 8 + wid; tile < n_tiles; tile += total_waves) {
    const long long row0 = (long long)tile * 16;

    // ---- A fragment: 16 points x 32 feats (two 32B chunks per lane) ----
    const float4* ar = (const float4*)(feats + (row0 + n) * MFEAT);
    const int kb = hi ? 2 : 0;  // float4 units: K base 0 or 8
    v16h a = pack_afrag(ar[kb], ar[kb + 1], ar[kb + 4], ar[kb + 5]);

    // ---- semantic head: [16x32] @ [32x20] via two WMMAs ----
    v8f c0 = {}; v8f c1 = {};
    c0 = WMMA_F16(a, bs0, c0);
    c1 = WMMA_F16(a, bs1, c1);
#pragma unroll
    for (int r = 0; r < 8; ++r) {
      slab[(r + m0) * 32 + n]      = c0[r] + sb0;
      slab[(r + m0) * 32 + n + 16] = c1[r] + sb1;
    }
    asm volatile("s_wait_dscnt 0x0" ::: "memory");

    if (lane < 16) {  // per-row argmax over 20 classes (first-max wins)
      const float* rowp = slab + lane * 32;
      float best = rowp[0]; int bi = 0;
#pragma unroll
      for (int c = 1; c < 20; ++c) { float v = rowp[c]; if (v > best) { best = v; bi = c; } }
      out_preds[row0 + lane] = (float)bi;
    }
#pragma unroll
    for (int it = 0; it < 10; ++it) {  // 320 contiguous floats, coalesced
      int idx = it * 32 + lane;
      int mm = idx / 20, cc = idx - mm * 20;
      out_scores[row0 * 20 + idx] = slab[mm * 32 + cc];
    }

    // ---- offset head layer 1: [16x32] @ [32x32], bias+BN+ReLU ----
    v8f h0 = {}; v8f h1 = {};
    h0 = WMMA_F16(a, b10, h0);
    h1 = WMMA_F16(a, b11, h1);
#pragma unroll
    for (int r = 0; r < 8; ++r) {
      slab[(r + m0) * 32 + n]      = fmaxf((h0[r] + ob0) * gg0 + bb0, 0.0f);
      slab[(r + m0) * 32 + n + 16] = fmaxf((h1[r] + ob1) * gg1 + bb1, 0.0f);
    }
    asm volatile("s_wait_dscnt 0x0" ::: "memory");

    // Reload h in A-fragment layout (ds_load_b128 x4) for the second GEMM.
    const float4* hr = (const float4*)(slab + n * 32 + (hi ? 8 : 0));
    v16h a2 = pack_afrag(hr[0], hr[1], hr[4], hr[5]);

    v8f d2 = {};
    d2 = WMMA_F16(a2, b2, d2);
    if (n < 3) {
#pragma unroll
      for (int r = 0; r < 8; ++r) slab[(r + m0) * 3 + n] = d2[r] + ob2;
    }
    asm volatile("s_wait_dscnt 0x0" ::: "memory");

    // 48 contiguous floats per tile, coalesced.
    out_offsets[row0 * 3 + lane] = slab[lane];
    if (lane < 16) out_offsets[row0 * 3 + 32 + lane] = slab[32 + lane];
  }
}

// ---------------------------------------------------------------------------
// Kernel B: zero the global pool workspace (ws is poisoned between runs).
// ---------------------------------------------------------------------------
__global__ void pg_init_pool(unsigned* __restrict__ p, int n) {
  int i = blockIdx.x * blockDim.x + threadIdx.x;
  if (i < n) p[i] = 0u;
}

// ---------------------------------------------------------------------------
// Kernel C: gather + BN + ReLU + segment-max. Full 1024x32 pool privatized
// in LDS (128KB of the WGP's 320KB). ReLU output >= 0 => unsigned bit-max ==
// float max, and init 0 reproduces the reference's empty-segment guard.
// One membership per wave-iteration: lanes load a full 128B feature row.
// ---------------------------------------------------------------------------
__global__ void __launch_bounds__(256) pg_gather_pool(
    const float* __restrict__ feats,
    const int*   __restrict__ pt_idx, const int* __restrict__ cl_idx,
    const float* __restrict__ g, const float* __restrict__ bt,
    unsigned* __restrict__ gpool, int T)
{
  __shared__ unsigned pool[NPROP * MFEAT];
  for (int i = threadIdx.x; i < NPROP * MFEAT; i += 256) pool[i] = 0u;
  __syncthreads();

  const int lane = threadIdx.x & 31;
  const int wid  = threadIdx.x >> 5;
  const float scale = g[lane] * rsqrtf(1.0f + BN_EPS);
  const float beta  = bt[lane];

  const int gw = (int)blockIdx.x * 8 + wid;
  const int nw = (int)gridDim.x * 8;

  for (long long base = (long long)gw * 32; base < T; base += (long long)nw * 32) {
    long long t = base + lane;
    int mpt = 0, mcl = 0;
    if (t < T) { mpt = pt_idx[t]; mcl = cl_idx[t]; }

    long long nb = base + (long long)nw * 32;   // prefetch next index chunk
    if (nb + lane < T) {
      __builtin_prefetch(pt_idx + nb + lane, 0, 0);
      __builtin_prefetch(cl_idx + nb + lane, 0, 0);
    }

    int cnt = (T - base < 32) ? (int)(T - base) : 32;
    for (int i = 0; i < cnt; ++i) {
      int pt = __shfl(mpt, i, 32);
      int cl = __shfl(mcl, i, 32);
      float v = feats[(long long)pt * MFEAT + lane];     // 128B coalesced, L2-hot
      v = fmaxf(v * scale + beta, 0.0f);
      atomicMax(&pool[cl * MFEAT + lane], __float_as_uint(v));  // ds_max_u32
    }
  }
  __syncthreads();

  for (int i = threadIdx.x; i < NPROP * MFEAT; i += 256) {
    unsigned v = pool[i];
    if (v) atomicMax(&gpool[i], v);   // global_atomic_max_u32, only touched slots
  }
}

// ---------------------------------------------------------------------------
// Kernel D: scores = pooled @ score_w + score_b  (1024 x 32 GEMV, tiny).
// ---------------------------------------------------------------------------
__global__ void pg_score_kernel(const unsigned* __restrict__ gpool,
                                const float* __restrict__ w,
                                const float* __restrict__ b,
                                float* __restrict__ out)
{
  int p = blockIdx.x * blockDim.x + threadIdx.x;
  if (p >= NPROP) return;
  float s = b[0];
#pragma unroll
  for (int f = 0; f < MFEAT; ++f)
    s += __uint_as_float(gpool[p * MFEAT + f]) * w[f];
  out[p] = s;
}

// ---------------------------------------------------------------------------
extern "C" void kernel_launch(void* const* d_in, const int* in_sizes, int n_in,
                              void* d_out, int out_size, void* d_ws, size_t ws_size,
                              hipStream_t stream)
{
  const float* feats   = (const float*)d_in[0];
  const float* sem_w   = (const float*)d_in[1];
  const float* sem_b   = (const float*)d_in[2];
  const float* off_w1  = (const float*)d_in[3];
  const float* off_b1  = (const float*)d_in[4];
  const float* bn_g    = (const float*)d_in[5];
  const float* bn_b    = (const float*)d_in[6];
  const float* off_w2  = (const float*)d_in[7];
  const float* off_b2  = (const float*)d_in[8];
  const float* sbn_g   = (const float*)d_in[9];
  const float* sbn_b   = (const float*)d_in[10];
  const float* score_w = (const float*)d_in[11];
  const float* score_b = (const float*)d_in[12];
  const int*   pt_idx  = (const int*)d_in[13];
  const int*   cl_idx  = (const int*)d_in[14];

  const long long N = (long long)in_sizes[0] / MFEAT;
  const int T = in_sizes[13];

  float* out         = (float*)d_out;
  float* out_scores  = out;
  float* out_preds   = out + N * 20;
  float* out_offsets = out + N * 21;
  float* out_prop    = out + N * 24;
  unsigned* gpool    = (unsigned*)d_ws;   // 1024*32*4B = 128KB scratch

  const int n_tiles = (int)(N / 16);      // N = 1e6 -> 62500 exact tiles

  pg_heads_kernel<<<1024, 256, 0, stream>>>(
      feats, sem_w, sem_b, off_w1, off_b1, bn_g, bn_b, off_w2, off_b2,
      out_scores, out_preds, out_offsets, n_tiles);

  pg_init_pool<<<(NPROP * MFEAT + 255) / 256, 256, 0, stream>>>(gpool, NPROP * MFEAT);

  pg_gather_pool<<<512, 256, 0, stream>>>(
      feats, pt_idx, cl_idx, sbn_g, sbn_b, gpool, T);

  pg_score_kernel<<<(NPROP + 255) / 256, 256, 0, stream>>>(
      gpool, score_w, score_b, out_prop);
}
